// BiasedConLoss_79723182948918
// MI455X (gfx1250) — matrix operand order repaired and checked by reading, not compiled, not run on hardware
//
#include <hip/hip_runtime.h>
#include <math.h>

// Problem constants (from reference): N=4096 rows per half, D=256, M=8192.
#define N_ROWS 4096
#define D_DIM  256
#define M_TOT  8192
#define NT     (M_TOT / 16)          // 512 column tiles
#define INV_T  (1.0f/0.07f)
#define LDS_STRIDE 264               // 256 + 8 bf16 pad (16B) per row

typedef __attribute__((ext_vector_type(16))) __bf16 v16bf;
typedef __attribute__((ext_vector_type(8)))  __bf16 v8bf;
typedef __attribute__((ext_vector_type(8)))  float  v8f;
typedef __attribute__((ext_vector_type(4)))  unsigned int u32x4;
typedef __attribute__((ext_vector_type(8)))  int i32x8;
typedef __attribute__((ext_vector_type(4)))  int i32x4;

union Frag16 { v16bf v; v8bf h[2]; };

#if defined(__HIP_DEVICE_COMPILE__) && __has_builtin(__builtin_amdgcn_tensor_load_to_lds)
#define USE_TDM 1
#else
#define USE_TDM 0
#endif

// ---------------- pass 0: split f32 features -> bf16 hi + bf16 lo ----------
__global__ __launch_bounds__(256)
void convert_kernel(const float* __restrict__ feat, const float* __restrict__ feat_cr,
                    __bf16* __restrict__ gHi, __bf16* __restrict__ gLo)
{
    const size_t idx = ((size_t)blockIdx.x * 256 + threadIdx.x) * 8;   // 8 elems/thread
    const size_t half = (size_t)N_ROWS * D_DIM;
    const float* src = (idx < half) ? (feat + idx) : (feat_cr + (idx - half));
    float4 x0 = *(const float4*)(src);
    float4 x1 = *(const float4*)(src + 4);
    float xs[8] = {x0.x, x0.y, x0.z, x0.w, x1.x, x1.y, x1.z, x1.w};
    v8bf hv, lv;
#pragma unroll
    for (int e = 0; e < 8; ++e) {
        __bf16 h = (__bf16)xs[e];
        hv[e] = h;
        lv[e] = (__bf16)(xs[e] - (float)h);
    }
    *(v8bf*)(gHi + idx) = hv;
    *(v8bf*)(gLo + idx) = lv;
}

#if USE_TDM
// LDS byte offset of a __shared__ object (generic -> as(3) -> integer).
__device__ __forceinline__ unsigned lds_off(const void* p) {
    return (unsigned)(unsigned long long)(__attribute__((address_space(3))) const void*)p;
}

// D# group0: count=1, lds_addr, 57-bit global addr, type=2 ("image").
__device__ __forceinline__ u32x4 tdm_group0(unsigned lds_addr, unsigned long long gaddr) {
    u32x4 g;
    g[0] = 1u;                                        // count=1, is_restore=0, no gather
    g[1] = lds_addr;                                  // [63:32]
    g[2] = (unsigned)(gaddr & 0xFFFFFFFFu);           // [95:64]
    g[3] = (unsigned)((gaddr >> 32) & 0x01FFFFFFu)    // [120:96] = addr[56:32]
         | (2u << 30);                                // [127:126] type=2
    return g;
}

// D# group1: 2B elems, pad 4 DWORDs (16B) every 128 DWORDs (512B row) ->
// hardware reproduces the 264-element padded LDS row stride.
// tensor: 8192 x 256, dim0 stride 256; tile: 16 x 256.
__device__ __forceinline__ i32x8 tdm_group1() {
    i32x8 g;
    g[0] = (1 << 16)      // data_size = 1 (2 bytes)
         | (1 << 20)      // pad_enable
         | (6 << 22)      // pad_interval: 2^(6+1)=128 DWORDs
         | (3 << 25);     // pad_amount : 3+1 = 4 DWORDs
    g[1] = (D_DIM & 0xFFFF) << 16;     // tensor_dim0[15:0] in [63:48]
    g[2] = (M_TOT & 0xFFFF) << 16;     // tensor_dim0 hi = 0; tensor_dim1 lo in [95:80]
    g[3] = (D_DIM & 0xFFFF) << 16;     // tensor_dim1 hi = 0; tile_dim0 in [127:112]
    g[4] = 16;                         // tile_dim1 = 16, tile_dim2 = 0
    g[5] = D_DIM;                      // tensor_dim0_stride[31:0]
    g[6] = 0;                          // stride hi, dim1_stride lo
    g[7] = 0;
    return g;
}

// This toolchain (clang-23 lane) declares the 6-arg form:
// (u32x4 g0, i32x8 g1, i32x4 g2, i32x4 g3, i32x8 extra, i32 cpol).
__device__ __forceinline__ void tdm_issue(u32x4 g0, i32x8 g1) {
    const i32x4 z4 = {0, 0, 0, 0};
    const i32x8 z8 = {0, 0, 0, 0, 0, 0, 0, 0};
    __builtin_amdgcn_tensor_load_to_lds(g0, g1, z4, z4, z8, 0);
}
#endif

// ---------------- pass 1: fused WMMA GEMM + online row statistics ----------
// 128 threads = 4 waves; wave owns 16 rows; 512 column tiles streamed through
// a double-buffered LDS staging area filled by the Tensor Data Mover.
__global__ __launch_bounds__(128)
void gemm_stats_kernel(const int*    __restrict__ labels,
                       const __bf16* __restrict__ gHi,
                       const __bf16* __restrict__ gLo,
                       float* __restrict__ wsMax, float* __restrict__ wsL,
                       float* __restrict__ wsPos, float* __restrict__ wsPart)
{
    __shared__ __bf16 sB[2][2][16 * LDS_STRIDE];   // [buf][hi/lo][rows*stride]
    __shared__ int    sLab[N_ROWS];

    const int tid  = threadIdx.x;
    const int wave = tid >> 5;
    const int lane = tid & 31;
    const int mloc = lane & 15;
    const int hh   = lane >> 4;
    const int rowBase = blockIdx.x * 64 + wave * 16;

    for (int i = tid; i < N_ROWS; i += 128) sLab[i] = labels[i];

    // A fragments straight from global bf16 (ISA 16-bit A 16x32 layout):
    // lane L: row L%16; elems[0..7] = K in [32c+8h,+8); elems[8..15] = +16.
    Frag16 aHi[8], aLo[8];
    {
        const __bf16* ah = gHi + (size_t)(rowBase + mloc) * D_DIM;
        const __bf16* al = gLo + (size_t)(rowBase + mloc) * D_DIM;
#pragma unroll
        for (int c = 0; c < 8; ++c) {
            const int k0 = 32 * c + 8 * hh;
            aHi[c].h[0] = *(const v8bf*)(ah + k0);
            aHi[c].h[1] = *(const v8bf*)(ah + k0 + 16);
            aLo[c].h[0] = *(const v8bf*)(al + k0);
            aLo[c].h[1] = *(const v8bf*)(al + k0 + 16);
        }
    }

    float accM[8], accL[8], accP[8], accQ[8];
#pragma unroll
    for (int v = 0; v < 8; ++v) { accM[v] = -INFINITY; accL[v] = 0.f; accP[v] = 0.f; accQ[v] = 0.f; }

#if USE_TDM
    const unsigned ldsHi0 = lds_off(&sB[0][0][0]), ldsHi1 = lds_off(&sB[1][0][0]);
    const unsigned ldsLo0 = lds_off(&sB[0][1][0]), ldsLo1 = lds_off(&sB[1][1][0]);
    const i32x8 g1 = tdm_group1();
    // Prologue: wave 0 launches the TDM for tile 0 into buffer 0.
    if (wave == 0) {
        tdm_issue(tdm_group0(ldsHi0, (unsigned long long)(const void*)gHi), g1);
        tdm_issue(tdm_group0(ldsLo0, (unsigned long long)(const void*)gLo), g1);
    }
#endif

    for (int jt = 0; jt < NT; ++jt) {
        const int cur = jt & 1;
        const int colBase = jt * 16;
        __syncthreads();   // everyone done reading buf[cur^1] before refill

#if USE_TDM
        if (wave == 0) {
            if (jt + 1 < NT) {
                // Kick tile jt+1 into the other buffer, then wait only for
                // tile jt (<=2 outstanding tensor ops): copy overlaps compute.
                const unsigned dHi = cur ? ldsHi0 : ldsHi1;
                const unsigned dLo = cur ? ldsLo0 : ldsLo1;
                const size_t off = (size_t)(colBase + 16) * D_DIM;
                tdm_issue(tdm_group0(dHi, (unsigned long long)(const void*)(gHi + off)), g1);
                tdm_issue(tdm_group0(dLo, (unsigned long long)(const void*)(gLo + off)), g1);
                __builtin_amdgcn_s_wait_tensorcnt(2);
            } else {
                __builtin_amdgcn_s_wait_tensorcnt(0);
            }
        }
#else
        // Fallback: manual bf16 copy (4 x 2KB passes, padded rows).
        for (int p = 0; p < 4; ++p) {
            const unsigned lin = p * 2048 + tid * 16;      // byte offset in 8KB tile
            const unsigned r   = lin >> 9;                 // 512B per row
            const unsigned wo  = lin & 511;
            const char* srcH = (const char*)(gHi + (size_t)colBase * D_DIM) + lin;
            const char* srcL = (const char*)(gLo + (size_t)colBase * D_DIM) + lin;
            uint4 dh = *(const uint4*)srcH;
            uint4 dl = *(const uint4*)srcL;
            *(uint4*)((char*)&sB[cur][0][0] + r * (LDS_STRIDE * 2) + wo) = dh;
            *(uint4*)((char*)&sB[cur][1][0] + r * (LDS_STRIDE * 2) + wo) = dl;
        }
#endif
        __syncthreads();

        // 16x16 tile, K=256, split-bf16: 3 independent WMMA chains.
        v8f acc0 = {0,0,0,0,0,0,0,0}, acc1 = acc0, acc2 = acc0;
        const __bf16* bhBase = &sB[cur][0][0] + mloc * LDS_STRIDE + 8 * hh;
        const __bf16* blBase = &sB[cur][1][0] + mloc * LDS_STRIDE + 8 * hh;
#pragma unroll
        for (int c = 0; c < 8; ++c) {
            Frag16 bh, bl;
            bh.h[0] = *(const v8bf*)(bhBase + 32 * c);
            bh.h[1] = *(const v8bf*)(bhBase + 32 * c + 16);
            bl.h[0] = *(const v8bf*)(blBase + 32 * c);
            bl.h[1] = *(const v8bf*)(blBase + 32 * c + 16);
            acc0 = __builtin_amdgcn_wmma_f32_16x16x32_bf16(false, aHi[c].v, false, bh.v,
                                                           (short)0, acc0, false, false);
            acc1 = __builtin_amdgcn_wmma_f32_16x16x32_bf16(false, aHi[c].v, false, bl.v,
                                                           (short)0, acc1, false, false);
            acc2 = __builtin_amdgcn_wmma_f32_16x16x32_bf16(false, aLo[c].v, false, bh.v,
                                                           (short)0, acc2, false, false);
        }

        // Online epilogue (C layout: lane L, VGPR v -> row v+8*hh, col L%16).
        const int col = colBase + mloc;
        const float isPos = (sLab[col & (N_ROWS - 1)] == 1) ? 1.f : 0.f;
#pragma unroll
        for (int v = 0; v < 8; ++v) {
            const int row = rowBase + v + 8 * hh;
            const float s = (acc0[v] + acc1[v] + acc2[v]) * INV_T;
            const bool diag   = (col == row);
            const bool pmatch = (col == (row ^ N_ROWS));   // (row+N)%M
            const float nm = fmaxf(accM[v], s);            // max includes diagonal
            const float e1 = __expf(accM[v] - nm);
            const float e2 = diag ? 0.f : __expf(s - nm);  // diagonal -> +1 at finalize
            accL[v] = accL[v] * e1 + e2;
            accM[v] = nm;
            accP[v] += (diag ? 0.f : isPos) * s;
            accQ[v] += pmatch ? s : 0.f;
        }
    }

    // Merge across the 16 lanes of each half (xor masks stay in-half).
#pragma unroll
    for (int off = 1; off < 16; off <<= 1) {
#pragma unroll
        for (int v = 0; v < 8; ++v) {
            const float om = __shfl_xor(accM[v], off, 32);
            const float ol = __shfl_xor(accL[v], off, 32);
            const float op = __shfl_xor(accP[v], off, 32);
            const float oq = __shfl_xor(accQ[v], off, 32);
            const float nm = fmaxf(accM[v], om);
            accL[v] = accL[v] * __expf(accM[v] - nm) + ol * __expf(om - nm);
            accM[v] = nm;
            accP[v] += op;
            accQ[v] += oq;
        }
    }
    if (mloc == 0) {
#pragma unroll
        for (int v = 0; v < 8; ++v) {
            const int row = rowBase + v + 8 * hh;
            wsMax[row]  = accM[v];
            wsL[row]    = accL[v];
            wsPos[row]  = accP[v];
            wsPart[row] = accQ[v];
        }
    }
}

// ---------------- pass 2: reduce 8192 per-row stats -> two scalars ---------
__global__ __launch_bounds__(256)
void finalize_kernel(const int* __restrict__ labels,
                     const float* __restrict__ wsMax, const float* __restrict__ wsL,
                     const float* __restrict__ wsPos, const float* __restrict__ wsPart,
                     float* __restrict__ out)
{
    __shared__ float red[256];
    __shared__ int   icnt[256];
    const int tid = threadIdx.x;

    int cnt = 0;
    for (int i = tid; i < N_ROWS; i += 256) cnt += (labels[i] == 1) ? 1 : 0;
    icnt[tid] = cnt;
    __syncthreads();
    for (int s = 128; s > 0; s >>= 1) { if (tid < s) icnt[tid] += icnt[tid + s]; __syncthreads(); }
    const float P = 2.0f * (float)icnt[0];
    const float U = 2.0f * (float)N_ROWS - P;

    float aSup = 0.f, aUns = 0.f;
    for (int i = tid; i < M_TOT; i += 256) {
        const int   lab  = labels[i & (N_ROWS - 1)];
        const float mm   = wsMax[i];
        const float lsum = logf(1.0f + wsL[i]);          // +1 = diagonal exp(0)
        if (lab == 1) {
            const float sup = (wsPos[i] - (P - 1.0f) * mm) - (float)M_TOT * lsum;
            aSup += -sup;
        } else {
            const float uns = (wsPart[i] - mm) - (float)M_TOT * lsum;
            aUns += -uns;
        }
    }
    red[tid] = aSup; __syncthreads();
    for (int s = 128; s > 0; s >>= 1) { if (tid < s) red[tid] += red[tid + s]; __syncthreads(); }
    const float totSup = red[0];
    __syncthreads();
    red[tid] = aUns; __syncthreads();
    for (int s = 128; s > 0; s >>= 1) { if (tid < s) red[tid] += red[tid + s]; __syncthreads(); }
    if (tid == 0) {
        out[0] = totSup / (P * P);
        out[1] = red[0] / (U * U);
    }
}

extern "C" void kernel_launch(void* const* d_in, const int* in_sizes, int n_in,
                              void* d_out, int out_size, void* d_ws, size_t ws_size,
                              hipStream_t stream)
{
    const int*   labels = (const int*)  d_in[0];
    const float* feat   = (const float*)d_in[1];
    const float* featcr = (const float*)d_in[2];

    // ws layout: 4x8192 f32 row stats (128KB), then bf16 hi (4MB) + lo (4MB).
    float*  wsMax  = (float*)d_ws;
    float*  wsL    = wsMax + M_TOT;
    float*  wsPos  = wsL   + M_TOT;
    float*  wsPart = wsPos + M_TOT;
    __bf16* gHi    = (__bf16*)((char*)d_ws + 131072);
    __bf16* gLo    = gHi + (size_t)M_TOT * D_DIM;

    convert_kernel<<<(M_TOT * D_DIM) / (256 * 8), 256, 0, stream>>>(feat, featcr, gHi, gLo);
    gemm_stats_kernel<<<M_TOT / 64, 128, 0, stream>>>(labels, gHi, gLo,
                                                      wsMax, wsL, wsPos, wsPart);
    finalize_kernel<<<1, 256, 0, stream>>>(labels, wsMax, wsL, wsPos, wsPart,
                                           (float*)d_out);
    (void)in_sizes; (void)n_in; (void)out_size; (void)ws_size;
}
